// Fusion_22643067585194
// MI455X (gfx1250) — compile-verified, matrix operand order checked by reading, not run on hardware
//
#include <hip/hip_runtime.h>
#include <hip/hip_bf16.h>

// ---------------------------------------------------------------------------
// Problem constants
// ---------------------------------------------------------------------------
#define NN   2
#define CC   144
#define HH   160
#define WW   160
#define HWv  (HH * WW)
#define CP   160          // NHWC padded channel count (multiple of 32)
#define EPSF 1e-5f
#define MID  64

typedef __attribute__((ext_vector_type(16))) __bf16 v16bf;
typedef __attribute__((ext_vector_type(8)))  float  v8f;

union Frag {
    v16bf  v;
    uint4  u[2];
    __bf16 e[16];
};

union PkBF {
    __bf16   h[2];
    unsigned u;
};

__device__ __forceinline__ float lrelu_f(float x) { return x > 0.f ? x : 0.2f * x; }
__device__ __forceinline__ float sigm_f(float x)  { return 1.f / (1.f + __expf(-x)); }

// ---------------------------------------------------------------------------
// Per-(n,c) mean / std over H*W  (std uses (n-1) denominator + EPS, sqrt)
// ---------------------------------------------------------------------------
__global__ void stats_kernel(const float* __restrict__ src,
                             float* __restrict__ omean, float* __restrict__ ostd)
{
    int c = blockIdx.x, n = blockIdx.y;
    const float* p = src + ((long)n * CC + c) * HWv;
    float s = 0.f, ss = 0.f;
    for (int i = threadIdx.x; i < HWv; i += blockDim.x) {
        float x = p[i];
        s += x; ss += x * x;
    }
    __shared__ float sh[256], sh2[256];
    sh[threadIdx.x] = s; sh2[threadIdx.x] = ss;
    __syncthreads();
    for (int st = 128; st > 0; st >>= 1) {
        if ((int)threadIdx.x < st) {
            sh[threadIdx.x]  += sh[threadIdx.x + st];
            sh2[threadIdx.x] += sh2[threadIdx.x + st];
        }
        __syncthreads();
    }
    if (threadIdx.x == 0) {
        float m   = sh[0] / (float)HWv;
        float var = (sh2[0] - sh[0] * sh[0] / (float)HWv) / (float)(HWv - 1) + EPSF;
        omean[n * CC + c] = m;
        ostd [n * CC + c] = sqrtf(var);
    }
}

// ---------------------------------------------------------------------------
// Tiny MLP: v(144) -> relu(64) -> relu(64) -> 1 scalar per batch
// ---------------------------------------------------------------------------
__global__ void mlp_kernel(const float* __restrict__ v,
                           const float* __restrict__ W1, const float* __restrict__ b1,
                           const float* __restrict__ W2, const float* __restrict__ b2,
                           const float* __restrict__ W3, const float* __restrict__ b3,
                           float* __restrict__ outp)
{
    int n = blockIdx.x;
    int j = threadIdx.x;                   // 64 threads
    __shared__ float h1[MID], h2[MID];
    float a = b1[j];
    for (int c = 0; c < CC; ++c) a += W1[j * CC + c] * v[n * CC + c];
    h1[j] = fmaxf(a, 0.f);
    __syncthreads();
    float b = b2[j];
    for (int k = 0; k < MID; ++k) b += W2[j * MID + k] * h1[k];
    h2[j] = fmaxf(b, 0.f);
    __syncthreads();
    if (j == 0) {
        float o = b3[0];
        for (int k = 0; k < MID; ++k) o += W3[k] * h2[k];
        outp[n] = o;
    }
}

// ---------------------------------------------------------------------------
// Fold AdaIN affine into the 1x1 conv: per-n scaled weights + folded bias.
// ---------------------------------------------------------------------------
__global__ void wscale_kernel(const float* __restrict__ w_alda, const float* __restrict__ b_alda,
                              const float* __restrict__ rmean,  const float* __restrict__ rstd,
                              const float* __restrict__ dstd_o, const float* __restrict__ dmean_o,
                              __bf16* __restrict__ wsc, float* __restrict__ bias2)
{
    int oc = blockIdx.x, n = blockIdx.y;
    int ci = threadIdx.x;                  // 160 threads
    float dstd = dstd_o[n], dmean = dmean_o[n];
    float wv = 0.f, partial = 0.f;
    if (ci < CC) {
        float a = dstd / rstd[n * CC + ci];
        float w = w_alda[oc * CC + ci];
        partial = w * (dmean - rmean[n * CC + ci] * a);
        wv = w * a;
    }
    wsc[((long)n * CC + oc) * CP + ci] = (__bf16)wv;
    __shared__ float sh[CP];
    sh[ci] = partial;
    __syncthreads();
    if (ci == 0) {
        float s = b_alda[oc];
        for (int k = 0; k < CP; ++k) s += sh[k];
        bias2[n * CC + oc] = s;
    }
}

// ---------------------------------------------------------------------------
// f32 NCHW -> bf16 NHWC (C padded to CP=160 with zeros)
// ---------------------------------------------------------------------------
__global__ void to_nhwc(const float* __restrict__ src, __bf16* __restrict__ dst)
{
    int w = blockIdx.x, h = blockIdx.y, n = blockIdx.z;
    int c = threadIdx.x;                   // 160 threads
    float x = (c < CC) ? src[(((long)n * CC + c) * HH + h) * WW + w] : 0.f;
    dst[(((long)n * HH + h) * WW + w) * CP + c] = (__bf16)x;
}

// Zero the padded channels [144,160) of the three intermediate NHWC buffers.
__global__ void pad_zero(__bf16* t0, __bf16* t1, __bf16* t2)
{
    int w = blockIdx.x, h = blockIdx.y, n = blockIdx.z;
    int buf = threadIdx.x >> 4, slot = CC + (threadIdx.x & 15);  // 48 threads
    __bf16* p = (buf == 0) ? t0 : (buf == 1) ? t1 : t2;
    p[(((long)n * HH + h) * WW + w) * CP + slot] = (__bf16)0.f;
}

// ---------------------------------------------------------------------------
// Weight repack: OIHW f32 -> bf16 rows [oc][tap][padded ci]
// ---------------------------------------------------------------------------
__global__ void wconv_kernel(const float* __restrict__ src, __bf16* __restrict__ dst, int dual)
{
    const int  KT     = dual ? 2 * CP : CP;
    const int  IC     = dual ? 2 * CC : CC;
    const long rowlen = 9L * KT;
    const long total  = (long)CC * rowlen;
    for (long idx = (long)blockIdx.x * blockDim.x + threadIdx.x; idx < total;
         idx += (long)gridDim.x * blockDim.x) {
        long oc = idx / rowlen;
        long r  = idx % rowlen;
        int  t  = (int)(r / KT);
        int  s  = (int)(r % KT);
        int  ci; bool valid;
        if (s < CP) { ci = s;             valid = (s < CC); }
        else        { ci = CC + (s - CP); valid = ((s - CP) < CC); }
        float x = valid ? src[(((long)oc * IC + ci) * 3 + t / 3) * 3 + (t % 3)] : 0.f;
        dst[idx] = (__bf16)x;
    }
}

// ---------------------------------------------------------------------------
// Implicit-GEMM conv via v_wmma_f32_16x16x32_bf16, fully specialized.
//   block = 288 threads = 9 waves; wave m owns oc tile [16m,16m+16)
//   grid  = (W/32, H, N): each block computes a 32-pixel row tile (two
//   16-px subtiles sharing A fragments), all 144 oc.
//   K is tap-major, per-tap padded to 160 channels (x NIN for concat),
//   so K chunks of 32 are exact and the unrolled inner loop has no branches.
// ---------------------------------------------------------------------------
template<int TAPS, int NIN, int ACT, bool DUAL, bool RESID, bool MUL, bool OUTF32, bool OUTBF>
__global__ __launch_bounds__(288)
void conv_wmma(const __bf16* __restrict__ act0, const __bf16* __restrict__ act1,
               const __bf16* __restrict__ wrow, long w_n_stride,
               const float* __restrict__ bias, int bias_n_stride,
               const __bf16* __restrict__ wrow2, const float* __restrict__ bias2,
               const float* __restrict__ resid, const float* __restrict__ mulsrc,
               float* __restrict__ out_f32, __bf16* __restrict__ out_bf)
{
    constexpr int KT = NIN * CP;

    const int lane = threadIdx.x & 31;
    const int m    = threadIdx.x >> 5;                 // 0..8
    const int n    = blockIdx.z, h = blockIdx.y, w0 = blockIdx.x * 32;
    const int px   = lane & 15;
    const int klh  = lane >> 4;
    const int kO0  = klh ? 8 : 0;                      // CDNA5 16-bit A/B K halves
    const int kO1  = klh ? 24 : 16;
    const long Krow = (long)TAPS * KT;
    const long wb   = (long)(m * 16 + px) * Krow + (long)n * w_n_stride;

    v8f acc[2]  = {};
    v8f acc2[2] = {};

    for (int t = 0; t < TAPS; ++t) {
        int hh, dxk;
        if (TAPS == 9) { hh = h + t / 3 - 1; dxk = t % 3 - 1; }
        else           { hh = h;             dxk = 0; }
        const int  wwp0 = w0 + px + dxk;
        const int  wwp1 = wwp0 + 16;
        const bool hok  = (hh >= 0) && (hh < HH);
        const bool in0  = hok && (wwp0 >= 0) && (wwp0 < WW);
        const bool in1  = hok && (wwp1 >= 0) && (wwp1 < WW);
        const long rowb = ((long)n * HH + hh) * WW;
        const long ab0  = (rowb + wwp0) * CP;
        const long ab1  = (rowb + wwp1) * CP;

#pragma unroll
        for (int c0 = 0; c0 < KT; c0 += 32) {
            Frag a, b0, b1;
            const long ka = wb + (long)t * KT + c0;
            __builtin_prefetch(wrow + ka + 256, 0, 1);  // global_prefetch
            a.u[0] = *(const uint4*)(wrow + ka + kO0);
            a.u[1] = *(const uint4*)(wrow + ka + kO1);

            const __bf16* src = (NIN == 2 && c0 >= CP) ? act1 : act0;
            const long    co  = (NIN == 2 && c0 >= CP) ? (long)(c0 - CP) : (long)c0;
            if (in0) {
                b0.u[0] = *(const uint4*)(src + ab0 + co + kO0);
                b0.u[1] = *(const uint4*)(src + ab0 + co + kO1);
            } else { b0.u[0] = make_uint4(0,0,0,0); b0.u[1] = make_uint4(0,0,0,0); }
            if (in1) {
                b1.u[0] = *(const uint4*)(src + ab1 + co + kO0);
                b1.u[1] = *(const uint4*)(src + ab1 + co + kO1);
            } else { b1.u[0] = make_uint4(0,0,0,0); b1.u[1] = make_uint4(0,0,0,0); }

            acc[0] = __builtin_amdgcn_wmma_f32_16x16x32_bf16(false, a.v, false, b0.v,
                                                             (short)0, acc[0], false, false);
            acc[1] = __builtin_amdgcn_wmma_f32_16x16x32_bf16(false, a.v, false, b1.v,
                                                             (short)0, acc[1], false, false);
            if (DUAL) {
                Frag a2;
                a2.u[0] = *(const uint4*)(wrow2 + ka + kO0);
                a2.u[1] = *(const uint4*)(wrow2 + ka + kO1);
                acc2[0] = __builtin_amdgcn_wmma_f32_16x16x32_bf16(false, a2.v, false, b0.v,
                                                                  (short)0, acc2[0], false, false);
                acc2[1] = __builtin_amdgcn_wmma_f32_16x16x32_bf16(false, a2.v, false, b1.v,
                                                                  (short)0, acc2[1], false, false);
            }
        }
    }

    // Epilogue: C/D layout — VGPR r: lanes 0-15 (M=r), lanes 16-31 (M=r+8)
    const int ocb = m * 16 + klh * 8;
#pragma unroll
    for (int s = 0; s < 2; ++s) {
        const int wcol = w0 + px + 16 * s;
        float v[8];
#pragma unroll
        for (int r = 0; r < 8; ++r) {
            const int oc = ocb + r;
            float x = acc[s][r] + bias[oc + n * bias_n_stride];
            if (DUAL) {
                float x2 = acc2[s][r] + bias2[oc];
                x = lrelu_f(x) * sigm_f(x2);
            } else if (ACT == 1) x = lrelu_f(x);
            else if (ACT == 2)   x = sigm_f(x);
            if (RESID || MUL || OUTF32) {
                const long nix = (((long)n * CC + oc) * HH + h) * WW + wcol;
                if (RESID)  x += resid[nix];
                if (MUL)    x *= mulsrc[nix];
                if (OUTF32) out_f32[nix] = x;
            }
            v[r] = x;
        }
        if (OUTBF) {
            __bf16* dp = out_bf + (((long)n * HH + h) * WW + wcol) * CP + ocb;
#pragma unroll
            for (int r = 0; r < 8; r += 2) {
                PkBF pk;
                pk.h[0] = (__bf16)v[r];
                pk.h[1] = (__bf16)v[r + 1];
                *(unsigned*)(dp + r) = pk.u;
            }
        }
    }
}

// ---------------------------------------------------------------------------
// Deformable conv (dg=8, 3x3) as WMMA implicit GEMM with bilinear-gathered B.
// ---------------------------------------------------------------------------
__global__ __launch_bounds__(288)
void dcn_wmma(const float* __restrict__ rgb, const float* __restrict__ off,
              const __bf16* __restrict__ wrow, __bf16* __restrict__ out_bf)
{
    const int lane = threadIdx.x & 31;
    const int m    = threadIdx.x >> 5;
    const int n    = blockIdx.z, h = blockIdx.y, w0 = blockIdx.x * 16;
    const int px   = lane & 15;
    const int klh  = lane >> 4;
    const int kO0  = klh ? 8 : 0;
    const int kO1  = klh ? 24 : 16;
    const long Krow = 9L * CP;
    const long wb   = (long)(m * 16 + px) * Krow;
    const int  wcol = w0 + px;

    v8f acc = {};

    for (int t = 0; t < 9; ++t) {
        const int ky = t / 3, kx = t % 3;
#pragma unroll
        for (int c0 = 0; c0 < CP; c0 += 32) {
            Frag a, b;
            const long ka = wb + (long)t * CP + c0;
            a.u[0] = *(const uint4*)(wrow + ka + kO0);
            a.u[1] = *(const uint4*)(wrow + ka + kO1);

            int   cur_dg = -1;
            float wgt[4];
            long  idx4[4];
#pragma unroll
            for (int e = 0; e < 16; ++e) {
                const int K = c0 + ((e < 8) ? (kO0 + e) : (kO1 + e - 8));
                if (K >= CC) { b.e[e] = (__bf16)0.f; continue; }
                const int ci = K;
                const int dg = ci / 18;
                if (dg != cur_dg) {
                    cur_dg = dg;
                    const long ob = ((long)n * CC + (dg * 9 + t) * 2) * (long)HWv
                                  + (long)h * WW + wcol;
                    const float dy = off[ob];
                    const float dx = off[ob + HWv];
                    const float py  = dy + (float)ky + (float)h - 1.f;
                    const float pxx = dx + (float)kx + (float)wcol - 1.f;
                    const float y0f = floorf(py), x0f = floorf(pxx);
                    const float wy = py - y0f, wx = pxx - x0f;
                    const int y0 = (int)y0f, x0 = (int)x0f;
#pragma unroll
                    for (int j = 0; j < 4; ++j) {
                        const int yc = y0 + (j >> 1), xc = x0 + (j & 1);
                        const bool valid = (yc >= 0) && (yc < HH) && (xc >= 0) && (xc < WW);
                        const int ycc = min(max(yc, 0), HH - 1);
                        const int xcc = min(max(xc, 0), WW - 1);
                        idx4[j] = (long)ycc * WW + xcc;
                        const float wj = ((j >> 1) ? wy : 1.f - wy) * ((j & 1) ? wx : 1.f - wx);
                        wgt[j] = valid ? wj : 0.f;
                    }
                }
                const float* base = rgb + ((long)n * CC + ci) * (long)HWv;
                const float val = wgt[0] * base[idx4[0]] + wgt[1] * base[idx4[1]]
                                + wgt[2] * base[idx4[2]] + wgt[3] * base[idx4[3]];
                b.e[e] = (__bf16)val;
            }
            acc = __builtin_amdgcn_wmma_f32_16x16x32_bf16(false, a.v, false, b.v,
                                                          (short)0, acc, false, false);
        }
    }

    const int ocb = m * 16 + klh * 8;
    __bf16* dp = out_bf + (((long)n * HH + h) * WW + wcol) * CP + ocb;
#pragma unroll
    for (int r = 0; r < 8; r += 2) {
        PkBF pk;
        pk.h[0] = (__bf16)acc[r];
        pk.h[1] = (__bf16)acc[r + 1];
        *(unsigned*)(dp + r) = pk.u;
    }
}

// ---------------------------------------------------------------------------
// Batchnorm over (N,H,W) per channel
// ---------------------------------------------------------------------------
__global__ void bn_stats(const float* __restrict__ gx,
                         float* __restrict__ bm, float* __restrict__ bv)
{
    int c = blockIdx.x;
    float s = 0.f, ss = 0.f;
    for (int n = 0; n < NN; ++n) {
        const float* p = gx + ((long)n * CC + c) * HWv;
        for (int i = threadIdx.x; i < HWv; i += blockDim.x) {
            float x = p[i];
            s += x; ss += x * x;
        }
    }
    __shared__ float sh[256], sh2[256];
    sh[threadIdx.x] = s; sh2[threadIdx.x] = ss;
    __syncthreads();
    for (int st = 128; st > 0; st >>= 1) {
        if ((int)threadIdx.x < st) {
            sh[threadIdx.x]  += sh[threadIdx.x + st];
            sh2[threadIdx.x] += sh2[threadIdx.x + st];
        }
        __syncthreads();
    }
    if (threadIdx.x == 0) {
        const float cnt = (float)(NN * HWv);
        float mn = sh[0] / cnt;
        bm[c] = mn;
        bv[c] = sh2[0] / cnt - mn * mn;
    }
}

__global__ void bn_apply(const float* __restrict__ gx,
                         const float* __restrict__ bm, const float* __restrict__ bv,
                         const float* __restrict__ g,  const float* __restrict__ b,
                         __bf16* __restrict__ dst)
{
    int w = blockIdx.x, h = blockIdx.y, n = blockIdx.z;
    int c = threadIdx.x;                   // 160 threads
    float x = 0.f;
    if (c < CC) {
        float xv = gx[(((long)n * CC + c) * HH + h) * WW + w];
        x = g[c] * (xv - bm[c]) * rsqrtf(bv[c] + EPSF) + b[c];
    }
    dst[(((long)n * HH + h) * WW + w) * CP + c] = (__bf16)x;
}

// ---------------------------------------------------------------------------
// Launch
// ---------------------------------------------------------------------------
extern "C" void kernel_launch(void* const* d_in, const int* in_sizes, int n_in,
                              void* d_out, int out_size, void* d_ws, size_t ws_size,
                              hipStream_t stream)
{
    (void)in_sizes; (void)n_in; (void)out_size; (void)ws_size;
    const float* depth   = (const float*)d_in[0];
    const float* rgb     = (const float*)d_in[1];
    const float* std_W1  = (const float*)d_in[2];
    const float* std_b1  = (const float*)d_in[3];
    const float* std_W2  = (const float*)d_in[4];
    const float* std_b2  = (const float*)d_in[5];
    const float* std_W3  = (const float*)d_in[6];
    const float* std_b3  = (const float*)d_in[7];
    const float* mean_W1 = (const float*)d_in[8];
    const float* mean_b1 = (const float*)d_in[9];
    const float* mean_W2 = (const float*)d_in[10];
    const float* mean_b2 = (const float*)d_in[11];
    const float* mean_W3 = (const float*)d_in[12];
    const float* mean_b3 = (const float*)d_in[13];
    const float* w_alda  = (const float*)d_in[14];
    const float* b_alda  = (const float*)d_in[15];
    const float* w_f1    = (const float*)d_in[16];
    const float* b_f1    = (const float*)d_in[17];
    const float* w_off   = (const float*)d_in[18];
    const float* b_off   = (const float*)d_in[19];
    const float* w_dcn   = (const float*)d_in[20];
    const float* w_f2    = (const float*)d_in[21];
    const float* b_f2    = (const float*)d_in[22];
    const float* w_gc    = (const float*)d_in[23];
    const float* b_gc    = (const float*)d_in[24];
    const float* w_gm    = (const float*)d_in[25];
    const float* b_gm    = (const float*)d_in[26];
    const float* bn_g    = (const float*)d_in[27];
    const float* bn_b    = (const float*)d_in[28];
    const float* w_p1    = (const float*)d_in[29];
    const float* b_p1    = (const float*)d_in[30];
    const float* w_p2    = (const float*)d_in[31];
    const float* b_p2    = (const float*)d_in[32];

    // ---- workspace layout -------------------------------------------------
    constexpr long SZ_NHWC = (long)NN * HH * WW * CP * 2;  // 16,384,000 B
    constexpr long SZ_F32T = (long)NN * CC * HH * WW * 4;  // 29,491,200 B
    constexpr long SZ_W1   = (long)CC * 9 * CP * 2;        //    414,720 B
    constexpr long SZ_W2   = (long)CC * 9 * 2 * CP * 2;    //    829,440 B

    char* ws = (char*)d_ws;
    long  o  = 0;
    auto take = [&](long sz) { long r = o; o += (sz + 255) & ~255L; return r; };

    __bf16* DEPTH_BF = (__bf16*)(ws + take(SZ_NHWC));
    __bf16* RGB_BF   = (__bf16*)(ws + take(SZ_NHWC));
    __bf16* T0       = (__bf16*)(ws + take(SZ_NHWC));  // rgb_lda -> feat_bn
    __bf16* T1       = (__bf16*)(ws + take(SZ_NHWC));  // off1 -> feat
    __bf16* T2       = (__bf16*)(ws + take(SZ_NHWC));  // rgb_guided -> p1
    float*  F0       = (float*) (ws + take(SZ_F32T));  // off -> gx
    __bf16* WF1      = (__bf16*)(ws + take(SZ_W2));
    __bf16* WOFF     = (__bf16*)(ws + take(SZ_W1));
    __bf16* WDCN     = (__bf16*)(ws + take(SZ_W1));
    __bf16* WF2      = (__bf16*)(ws + take(SZ_W2));
    __bf16* WGC      = (__bf16*)(ws + take(SZ_W1));
    __bf16* WGM      = (__bf16*)(ws + take(SZ_W1));
    __bf16* WP1      = (__bf16*)(ws + take(SZ_W2));
    __bf16* WP2      = (__bf16*)(ws + take(SZ_W1));
    __bf16* WSC      = (__bf16*)(ws + take((long)NN * CC * CP * 2));
    float*  BIAS2    = (float*) (ws + take((long)NN * CC * 4));
    float*  DMEAN    = (float*) (ws + take((long)NN * CC * 4));
    float*  DSTD     = (float*) (ws + take((long)NN * CC * 4));
    float*  RMEAN    = (float*) (ws + take((long)NN * CC * 4));
    float*  RSTD     = (float*) (ws + take((long)NN * CC * 4));
    float*  MLP_STD  = (float*) (ws + take(NN * 4));
    float*  MLP_MEAN = (float*) (ws + take(NN * 4));
    float*  BNM      = (float*) (ws + take(CC * 4));
    float*  BNV      = (float*) (ws + take(CC * 4));

    const dim3 gHW(WW, HH, NN);
    const dim3 gConv(WW / 32, HH, NN);   // 32-pixel tiles
    const dim3 gDcn(WW / 16, HH, NN);

    // ---- prep -------------------------------------------------------------
    to_nhwc<<<gHW, 160, 0, stream>>>(depth, DEPTH_BF);
    to_nhwc<<<gHW, 160, 0, stream>>>(rgb,   RGB_BF);
    pad_zero<<<gHW, 48, 0, stream>>>(T0, T1, T2);

    wconv_kernel<<<512, 256, 0, stream>>>(w_f1,  WF1, 1);
    wconv_kernel<<<512, 256, 0, stream>>>(w_off, WOFF, 0);
    wconv_kernel<<<512, 256, 0, stream>>>(w_dcn, WDCN, 0);
    wconv_kernel<<<512, 256, 0, stream>>>(w_f2,  WF2, 1);
    wconv_kernel<<<512, 256, 0, stream>>>(w_gc,  WGC, 0);
    wconv_kernel<<<512, 256, 0, stream>>>(w_gm,  WGM, 0);
    wconv_kernel<<<512, 256, 0, stream>>>(w_p1,  WP1, 1);
    wconv_kernel<<<512, 256, 0, stream>>>(w_p2,  WP2, 0);

    stats_kernel<<<dim3(CC, NN), 256, 0, stream>>>(depth, DMEAN, DSTD);
    stats_kernel<<<dim3(CC, NN), 256, 0, stream>>>(rgb,   RMEAN, RSTD);

    mlp_kernel<<<NN, MID, 0, stream>>>(DSTD,  std_W1,  std_b1,  std_W2,  std_b2,  std_W3,  std_b3,  MLP_STD);
    mlp_kernel<<<NN, MID, 0, stream>>>(DMEAN, mean_W1, mean_b1, mean_W2, mean_b2, mean_W3, mean_b3, MLP_MEAN);

    wscale_kernel<<<dim3(CC, NN), CP, 0, stream>>>(w_alda, b_alda, RMEAN, RSTD,
                                                   MLP_STD, MLP_MEAN, WSC, BIAS2);

    // ---- main pipeline (all WMMA) -----------------------------------------
    // rgb_lda = rgb + conv1x1(folded affine)           -> T0 (bf16 NHWC)
    conv_wmma<1, 1, 0, false, true, false, false, true>
        <<<gConv, 288, 0, stream>>>(RGB_BF, nullptr, WSC, (long)CC * CP, BIAS2, CC,
                                    nullptr, nullptr, rgb, nullptr, nullptr, T0);
    // off1 = lrelu(conv3x3(concat(depth, rgb_lda)))    -> T1
    conv_wmma<9, 2, 1, false, false, false, false, true>
        <<<gConv, 288, 0, stream>>>(DEPTH_BF, T0, WF1, 0, b_f1, 0,
                                    nullptr, nullptr, nullptr, nullptr, nullptr, T1);
    // off = lrelu(conv3x3(off1))                       -> F0 (f32 NCHW)
    conv_wmma<9, 1, 1, false, false, false, true, false>
        <<<gConv, 288, 0, stream>>>(T1, nullptr, WOFF, 0, b_off, 0,
                                    nullptr, nullptr, nullptr, nullptr, F0, nullptr);
    // rgb_guided = deform_conv(rgb, off, w_dcn)        -> T2
    dcn_wmma<<<gDcn, 288, 0, stream>>>(rgb, F0, WDCN, T2);
    // feat = conv3x3(concat(depth, rgb_guided))        -> T1
    conv_wmma<9, 2, 0, false, false, false, false, true>
        <<<gConv, 288, 0, stream>>>(DEPTH_BF, T2, WF2, 0, b_f2, 0,
                                    nullptr, nullptr, nullptr, nullptr, nullptr, T1);
    // gx = lrelu(conv_gc(feat)) * sigmoid(conv_gm(feat)) -> F0 (f32 NCHW)
    conv_wmma<9, 1, 0, true, false, false, true, false>
        <<<gConv, 288, 0, stream>>>(T1, nullptr, WGC, 0, b_gc, 0,
                                    WGM, b_gm, nullptr, nullptr, F0, nullptr);
    // feat = batchnorm(gx)                              -> T0
    bn_stats<<<CC, 256, 0, stream>>>(F0, BNM, BNV);
    bn_apply<<<gHW, 160, 0, stream>>>(F0, BNM, BNV, bn_g, bn_b, T0);
    // p1 = lrelu(conv3x3(concat(depth, feat)))          -> T2
    conv_wmma<9, 2, 1, false, false, false, false, true>
        <<<gConv, 288, 0, stream>>>(DEPTH_BF, T0, WP1, 0, b_p1, 0,
                                    nullptr, nullptr, nullptr, nullptr, nullptr, T2);
    // out = depth * sigmoid(conv3x3(p1))                -> d_out (f32 NCHW)
    conv_wmma<9, 1, 2, false, false, true, true, false>
        <<<gConv, 288, 0, stream>>>(T2, nullptr, WP2, 0, b_p2, 0,
                                    nullptr, nullptr, nullptr, depth, (float*)d_out, nullptr);
}
